// Hgru2ScalarDecay_50586124812309
// MI455X (gfx1250) — compile-verified, matrix operand order checked by reading, not run on hardware
//
#include <hip/hip_runtime.h>
#include <hip/hip_bf16.h>
#include <stdint.h>

// ---------------------------------------------------------------------------
// Types for CDNA5 WMMA (wave32): 16x16x32 bf16 -> f32
// ---------------------------------------------------------------------------
typedef __attribute__((ext_vector_type(16))) __bf16 bf16x16;
typedef __attribute__((ext_vector_type(8)))  __bf16 bf16x8;
typedef __attribute__((ext_vector_type(8)))  float  f32x8;
typedef __attribute__((ext_vector_type(4)))  int    i32x4;

#define D_   2048
#define H_   16
#define E_   128
#define C_   64
#define MROWS 4096   // b*n = 2*2048

// ---------------------------------------------------------------------------
// Async global->LDS path (gfx1250 GLOBAL_LOAD_ASYNC_TO_LDS_B128, ASYNCcnt).
// Builtin signature (from compiler diagnostic): (as1 v4i*, as3 v4i*, imm, imm)
// ---------------------------------------------------------------------------
#if defined(__AMDGCN__) && __has_builtin(__builtin_amdgcn_global_load_async_to_lds_b128) && __has_builtin(__builtin_amdgcn_s_wait_asynccnt)
#define HAS_ASYNC_LDS 1
#else
#define HAS_ASYNC_LDS 0
#endif

typedef __attribute__((address_space(1))) i32x4* as1v4p;
typedef __attribute__((address_space(3))) i32x4* as3v4p;

__device__ __forceinline__ void ld_async16(__bf16* lds, const __bf16* g) {
#if HAS_ASYNC_LDS
  __builtin_amdgcn_global_load_async_to_lds_b128(
      (as1v4p)const_cast<__bf16*>(g), (as3v4p)lds, 0, 0);
#else
  *(bf16x8*)lds = *(const bf16x8*)g;
#endif
}
__device__ __forceinline__ void wait_async() {
#if HAS_ASYNC_LDS
  __builtin_amdgcn_s_wait_asynccnt(0);
#endif
}

__device__ __forceinline__ f32x8 zero8() {
  f32x8 z = {0.f,0.f,0.f,0.f,0.f,0.f,0.f,0.f}; return z;
}
__device__ __forceinline__ f32x8 wmma_bf(bf16x16 a, bf16x16 b, f32x8 c) {
  return __builtin_amdgcn_wmma_f32_16x16x32_bf16(false, a, false, b, (short)0, c, false, false);
}

// ---------------------------------------------------------------------------
// Fragment helpers (ISA 7.12.2 layouts, wave32). Native bf16 converts only.
// ---------------------------------------------------------------------------
__device__ __forceinline__ bf16x16 load_frag(const __bf16* p, int ldv) {
  int lane = threadIdx.x & 31;
  int r  = lane & 15;
  int hh = (lane >> 4) << 3;   // 0 or 8
  const __bf16* q0 = p + (size_t)r * ldv + hh;
  union { bf16x8 v8[2]; bf16x16 v16; } u;
  u.v8[0] = *(const bf16x8*)(q0);
  u.v8[1] = *(const bf16x8*)(q0 + 16);
  return u.v16;
}
// per-lane row scale in bf16 (rows are lane-uniform for A fragments)
__device__ __forceinline__ bf16x16 scale_frag(bf16x16 a, float s) {
  __bf16 sb = (__bf16)s;
  bf16x16 o;
#pragma unroll
  for (int i = 0; i < 16; ++i) o[i] = a[i] * sb;
  return o;
}
// stack two vertically adjacent C tiles into one 32x16 bf16 B operand
__device__ __forceinline__ bf16x16 pack_b_from_c(f32x8 c0, f32x8 c1) {
  bf16x16 o;
#pragma unroll
  for (int i = 0; i < 8; ++i) { o[i] = (__bf16)c0[i]; o[8 + i] = (__bf16)c1[i]; }
  return o;
}
__device__ __forceinline__ void store_c_f32(float* p, int ldv, f32x8 c) {
  int lane = threadIdx.x & 31;
  int n  = lane & 15;
  int mb = (lane >> 4) << 3;
#pragma unroll
  for (int r = 0; r < 8; ++r) p[(size_t)(mb + r) * ldv + n] = c[r];
}
__device__ __forceinline__ void store_c_bf16(__bf16* p, int ldv, f32x8 c) {
  int lane = threadIdx.x & 31;
  int n  = lane & 15;
  int mb = (lane >> 4) << 3;
#pragma unroll
  for (int r = 0; r < 8; ++r) p[(size_t)(mb + r) * ldv + n] = (__bf16)c[r];
}

// ---------------------------------------------------------------------------
// fp32 -> bf16 cast
// ---------------------------------------------------------------------------
__global__ void cast_bf16_kernel(const float* __restrict__ in,
                                 __bf16* __restrict__ out, int n) {
  int i = blockIdx.x * blockDim.x + threadIdx.x;
  if (i < n) out[i] = (__bf16)in[i];
}

// ---------------------------------------------------------------------------
// Tiled bf16 WMMA GEMM: C[M,N] = act(A[M,K] @ B[K,N])
// 256 threads (8 waves), block tile 128x128, wave tile 64x32, K step 32.
// Double-buffered LDS staging: A tiles go through the async global->LDS path
// (overlapped with WMMA of the current tile), B is transposed n-major so both
// operand fragments are conflict-free 16B ds_load_b128s (rows padded to 40).
// ---------------------------------------------------------------------------
#define ALD 40
template<int ACT, int OUTBF>   // ACT: 0 none, 1 silu ; OUTBF: 1 bf16 out, 0 f32 out
__global__ __launch_bounds__(256) void gemm_bf16_kernel(
    const __bf16* __restrict__ A, const __bf16* __restrict__ B,
    float* __restrict__ Cf, __bf16* __restrict__ Cb,
    int M, int N, int K) {
  __shared__ __align__(16) __bf16 As[2][128 * ALD];
  __shared__ __align__(16) __bf16 Bs[2][128 * ALD];   // transposed: Bs[n][k]

  const int t    = threadIdx.x;
  const int wave = t >> 5;
  const int wm   = wave & 1;   // 2 waves along M
  const int wn   = wave >> 1;  // 4 waves along N
  const int m0   = blockIdx.y * 128;
  const int n0   = blockIdx.x * 128;

  f32x8 acc[4][2];
#pragma unroll
  for (int i = 0; i < 4; ++i)
#pragma unroll
    for (int j = 0; j < 2; ++j) acc[i][j] = zero8();

  const int tra = t >> 1, tca = (t & 1) * 16;   // A stage map (128x32)
  const int krb = t >> 3, ncb = (t & 7) * 16;   // B stage map (32x128)

  auto stage = [&](int buf, int k0) {
    const __bf16* ga = A + (size_t)(m0 + tra) * K + k0 + tca;
    ld_async16(&As[buf][tra * ALD + tca], ga);
    ld_async16(&As[buf][tra * ALD + tca + 8], ga + 8);
    const __bf16* gb = B + (size_t)(k0 + krb) * N + n0 + ncb;
    bf16x8 b0 = *(const bf16x8*)(gb);
    bf16x8 b1 = *(const bf16x8*)(gb + 8);
#pragma unroll
    for (int i = 0; i < 8; ++i) {
      Bs[buf][(ncb + i)     * ALD + krb] = b0[i];
      Bs[buf][(ncb + 8 + i) * ALD + krb] = b1[i];
    }
    __builtin_prefetch(gb + (size_t)32 * N, 0, 3);   // global_prefetch_b8
  };

  const int KT = K >> 5;
  stage(0, 0);
  wait_async();
  __syncthreads();

  for (int kt = 0; kt < KT; ++kt) {
    const int cur = kt & 1;
    if (kt + 1 < KT) stage(cur ^ 1, (kt + 1) * 32);   // overlap with WMMA below

    bf16x16 af[4], bfg[2];
#pragma unroll
    for (int i = 0; i < 4; ++i) af[i]  = load_frag(&As[cur][(wm * 64 + 16 * i) * ALD], ALD);
#pragma unroll
    for (int j = 0; j < 2; ++j) bfg[j] = load_frag(&Bs[cur][(wn * 32 + 16 * j) * ALD], ALD);
#pragma unroll
    for (int i = 0; i < 4; ++i)
#pragma unroll
      for (int j = 0; j < 2; ++j) acc[i][j] = wmma_bf(af[i], bfg[j], acc[i][j]);

    wait_async();
    __syncthreads();
  }

#pragma unroll
  for (int i = 0; i < 4; ++i)
#pragma unroll
    for (int j = 0; j < 2; ++j) {
      f32x8 c = acc[i][j];
      if (ACT == 1) {
#pragma unroll
        for (int e = 0; e < 8; ++e) { float x = c[e]; c[e] = x / (1.f + __expf(-x)); }
      }
      const size_t base = (size_t)(m0 + wm * 64 + 16 * i) * N + (n0 + wn * 32 + 16 * j);
      if (OUTBF) store_c_bf16(Cb + base, N, c);
      else       store_c_f32 (Cf + base, N, c);
    }
}

// ---------------------------------------------------------------------------
// ld = log_sigmoid(x @ Wf) : (4096,16) outputs, one thread each
// ---------------------------------------------------------------------------
__global__ __launch_bounds__(256) void ld_kernel(const float* __restrict__ x,
                                                 const float* __restrict__ Wf,
                                                 float* __restrict__ out) {
  int gid = blockIdx.x * blockDim.x + threadIdx.x;   // 0..65535
  int row = gid >> 4, hh = gid & 15;
  const float* xr = x + (size_t)row * D_;
  float acc = 0.f;
  for (int d = 0; d < D_; ++d) acc += xr[d] * Wf[(size_t)d * H_ + hh];
  out[gid] = fminf(acc, 0.f) - log1pf(__expf(-fabsf(acc)));
}

// ---------------------------------------------------------------------------
// Chunked lightning attention with scalar decay.
// One workgroup (128 threads = 4 waves) per (batch, head). The fp32 recurrent
// state S (128x128) lives in WMMA accumulator fragments (wave w owns all 8
// e-tiles of its 2 f-tiles). S -> bf16 B operand for q@S is a pure per-lane
// repack (C-tile stacking). kd is written TRANSPOSED into an LDS buffer that
// overlays the (then dead) qs/ss region, so the S-update A fragments are
// contiguous 16B ds_load_b128s instead of 16-way u16 gathers.
// ---------------------------------------------------------------------------
#define QLD 136   // q/k chunk LDS row stride (elems); 272B rows, 16B aligned
#define VLD 72    // v^T / kd^T / scores LDS row stride
#define SLD 72
#define KDLD 72
// carved offsets inside one shared block (elems):
//   qs : [0, 8704)      64 x QLD
//   ss : [8704, 13312)  64 x SLD
//   ks : [13312, 22016) 64 x QLD
//   vts: [22016, 31232) 128 x VLD
//   kdt: [0, 9216)      128 x KDLD   (overlays qs+ss after o is computed)
__global__ __launch_bounds__(128) void attn_kernel(
    const __bf16* __restrict__ qg, const __bf16* __restrict__ kg,
    const __bf16* __restrict__ vg, const float*  __restrict__ ldg,
    float* __restrict__ og) {
  __shared__ __align__(16) __bf16 smem[31232];
  __shared__ float cum[64], eq[64], eqi[64], edec[64];
  __shared__ float tot_s;
  __bf16* qs  = smem;
  __bf16* ss  = smem + 8704;
  __bf16* ks  = smem + 13312;
  __bf16* vts = smem + 22016;
  __bf16* kdt = smem;

  const int t    = threadIdx.x;
  const int lane = t & 31;
  const int wave = t >> 5;
  const int bidx = blockIdx.x >> 4;   // / H
  const int h    = blockIdx.x & 15;

  f32x8 S[8][2];
#pragma unroll
  for (int e = 0; e < 8; ++e)
#pragma unroll
    for (int j = 0; j < 2; ++j) S[e][j] = zero8();

  for (int c = 0; c < 32; ++c) {
    const int rb = bidx * 2048 + c * C_;   // global row base of this chunk
    __syncthreads();   // previous chunk's LDS consumers done

    // ---- stage q, k (async, row-major) and v (transposed) ---------------
#pragma unroll
    for (int rep = 0; rep < 8; ++rep) {
      int e2 = (rep * 128 + t) * 8;
      int r = e2 >> 7, cc = e2 & 127;
      const size_t gofs = (size_t)(rb + r) * D_ + h * E_ + cc;
      ld_async16(&qs[r * QLD + cc], qg + gofs);
      ld_async16(&ks[r * QLD + cc], kg + gofs);
      bf16x8 vv = *(const bf16x8*)(vg + gofs);
#pragma unroll
      for (int j2 = 0; j2 < 8; ++j2) vts[(cc + j2) * VLD + r] = vv[j2];
    }
    if (t == 0) {
      float run = 0.f;
      for (int i = 0; i < C_; ++i) { run += ldg[(size_t)(rb + i) * H_ + h]; cum[i] = run; }
      tot_s = run;
    }
    wait_async();
    __syncthreads();
    if (t < 64) {
      float e = __expf(cum[t]);
      float inv = 1.f / e;                 // exp(-cum)
      eq[t] = e; eqi[t] = inv;
      edec[t] = __expf(tot_s) * inv;       // exp(tot - cum)
    }
    __syncthreads();

    // ---- intra scores: wave handles m-tile = wave ------------------------
    {
      const int mt = wave;
      const int nloc = lane & 15, mb = (lane >> 4) << 3;
      for (int j = 0; j <= mt; ++j) {
        f32x8 sc = zero8();
#pragma unroll
        for (int s = 0; s < 4; ++s) {
          bf16x16 a = load_frag(&qs[(16 * mt) * QLD + 32 * s], QLD);
          bf16x16 b = load_frag(&ks[(16 * j)  * QLD + 32 * s], QLD);  // B = k^T
          sc = wmma_bf(a, b, sc);
        }
#pragma unroll
        for (int r = 0; r < 8; ++r) {
          int m = 16 * mt + mb + r, n = 16 * j + nloc;
          float val = (m >= n) ? sc[r] * (eq[m] * eqi[n]) : 0.f;
          ss[m * SLD + n] = (__bf16)val;
        }
      }
      for (int j = mt + 1; j < 4; ++j) {   // strictly-upper tiles are zero
#pragma unroll
        for (int r = 0; r < 8; ++r)
          ss[(16 * mt + mb + r) * SLD + 16 * j + nloc] = (__bf16)0.f;
      }
    }
    __syncthreads();   // scores visible

    // ---- o = scores @ v + (q*exp(cum)) @ S ; wave owns f-tiles 2w,2w+1 ---
    f32x8 accO[4][2];
#pragma unroll
    for (int i = 0; i < 4; ++i)
#pragma unroll
      for (int j = 0; j < 2; ++j) accO[i][j] = zero8();

#pragma unroll
    for (int s = 0; s < 2; ++s) {           // intra, K = 64
      bf16x16 bv[2];
#pragma unroll
      for (int jf = 0; jf < 2; ++jf)
        bv[jf] = load_frag(&vts[(16 * (2 * wave + jf)) * VLD + 32 * s], VLD);
#pragma unroll
      for (int mt = 0; mt < 4; ++mt) {
        bf16x16 a = load_frag(&ss[(16 * mt) * SLD + 32 * s], SLD);
#pragma unroll
        for (int jf = 0; jf < 2; ++jf) accO[mt][jf] = wmma_bf(a, bv[jf], accO[mt][jf]);
      }
    }
#pragma unroll
    for (int s = 0; s < 4; ++s) {           // inter, K = 128 (uses OLD S)
      bf16x16 bS[2];
#pragma unroll
      for (int jf = 0; jf < 2; ++jf) bS[jf] = pack_b_from_c(S[2 * s][jf], S[2 * s + 1][jf]);
#pragma unroll
      for (int mt = 0; mt < 4; ++mt) {
        float qsc = eq[16 * mt + (lane & 15)];
        bf16x16 a = scale_frag(load_frag(&qs[(16 * mt) * QLD + 32 * s], QLD), qsc);
#pragma unroll
        for (int jf = 0; jf < 2; ++jf) accO[mt][jf] = wmma_bf(a, bS[jf], accO[mt][jf]);
      }
    }
#pragma unroll
    for (int mt = 0; mt < 4; ++mt)
#pragma unroll
      for (int jf = 0; jf < 2; ++jf)
        store_c_f32(og + (size_t)(rb + 16 * mt) * D_ + h * E_ + 16 * (2 * wave + jf),
                    D_, accO[mt][jf]);

    __syncthreads();   // all qs/ss reads complete -> kdt may overlay them

    // ---- kd^T = (k * exp(tot - cum[i]))^T into kdt ----------------------
#pragma unroll
    for (int rep = 0; rep < 8; ++rep) {
      int e2 = (rep * 128 + t) * 8;
      int r = e2 >> 7, cc = e2 & 127;
      __bf16 scl = (__bf16)edec[r];
      bf16x8 kv = *(const bf16x8*)&ks[r * QLD + cc];
#pragma unroll
      for (int j2 = 0; j2 < 8; ++j2) kdt[(cc + j2) * KDLD + r] = kv[j2] * scl;
    }
    __syncthreads();   // kdt visible

    // ---- S = S * exp(tot) + kd^T @ v ------------------------------------
    const float etot = __expf(tot_s);
#pragma unroll
    for (int e = 0; e < 8; ++e)
#pragma unroll
      for (int jf = 0; jf < 2; ++jf)
#pragma unroll
        for (int i = 0; i < 8; ++i) S[e][jf][i] *= etot;

#pragma unroll
    for (int si = 0; si < 2; ++si) {        // K = 64 over chunk index i
      bf16x16 bv[2];
#pragma unroll
      for (int jf = 0; jf < 2; ++jf)
        bv[jf] = load_frag(&vts[(16 * (2 * wave + jf)) * VLD + 32 * si], VLD);
#pragma unroll
      for (int et = 0; et < 8; ++et) {
        bf16x16 a = load_frag(&kdt[(16 * et) * KDLD + 32 * si], KDLD);  // kd^T
#pragma unroll
        for (int jf = 0; jf < 2; ++jf) S[et][jf] = wmma_bf(a, bv[jf], S[et][jf]);
      }
    }
  }
}

// ---------------------------------------------------------------------------
// RMS norm over last dim (2048) -> bf16 for the final GEMM
// ---------------------------------------------------------------------------
__global__ __launch_bounds__(256) void rms_kernel(const float* __restrict__ o,
                                                  const float* __restrict__ nw,
                                                  __bf16* __restrict__ out) {
  __shared__ float red[256];
  const int row = blockIdx.x;
  const float* r = o + (size_t)row * D_;
  float s = 0.f;
  for (int i = threadIdx.x; i < D_; i += 256) { float v = r[i]; s += v * v; }
  red[threadIdx.x] = s;
  __syncthreads();
  for (int st = 128; st > 0; st >>= 1) {
    if (threadIdx.x < st) red[threadIdx.x] += red[threadIdx.x + st];
    __syncthreads();
  }
  const float scale = rsqrtf(red[0] / (float)D_ + 1e-6f);
  for (int i = threadIdx.x; i < D_; i += 256)
    out[(size_t)row * D_ + i] = (__bf16)(r[i] * scale * nw[i]);
}

// ---------------------------------------------------------------------------
// Launch
// ---------------------------------------------------------------------------
extern "C" void kernel_launch(void* const* d_in, const int* in_sizes, int n_in,
                              void* d_out, int out_size, void* d_ws, size_t ws_size,
                              hipStream_t stream) {
  const float* x  = (const float*)d_in[0];
  const float* Wq = (const float*)d_in[1];
  const float* Wk = (const float*)d_in[2];
  const float* Wv = (const float*)d_in[3];
  const float* Wf = (const float*)d_in[4];
  const float* Wo = (const float*)d_in[5];
  const float* nw = (const float*)d_in[6];
  float* out = (float*)d_out;

  size_t off = 0;
  auto carve = [&](size_t bytes) -> void* {
    void* r = (char*)d_ws + off;
    off += (bytes + 255) & ~(size_t)255;
    return r;
  };
  __bf16* xb  = (__bf16*)carve((size_t)MROWS * D_ * 2);
  __bf16* wqb = (__bf16*)carve((size_t)D_ * D_ * 2);
  __bf16* wkb = (__bf16*)carve((size_t)D_ * D_ * 2);
  __bf16* wvb = (__bf16*)carve((size_t)D_ * D_ * 2);
  __bf16* wob = (__bf16*)carve((size_t)D_ * D_ * 2);
  __bf16* qb  = (__bf16*)carve((size_t)MROWS * D_ * 2);
  __bf16* kb  = (__bf16*)carve((size_t)MROWS * D_ * 2);
  __bf16* vb  = (__bf16*)carve((size_t)MROWS * D_ * 2);
  float*  ldb = (float*) carve((size_t)MROWS * H_ * 4);
  float*  ob  = (float*) carve((size_t)MROWS * D_ * 4);
  __bf16* onb = (__bf16*)carve((size_t)MROWS * D_ * 2);

  const int nX = MROWS * D_;
  const int nW = D_ * D_;
  cast_bf16_kernel<<<(nX + 255) / 256, 256, 0, stream>>>(x,  xb,  nX);
  cast_bf16_kernel<<<(nW + 255) / 256, 256, 0, stream>>>(Wq, wqb, nW);
  cast_bf16_kernel<<<(nW + 255) / 256, 256, 0, stream>>>(Wk, wkb, nW);
  cast_bf16_kernel<<<(nW + 255) / 256, 256, 0, stream>>>(Wv, wvb, nW);
  cast_bf16_kernel<<<(nW + 255) / 256, 256, 0, stream>>>(Wo, wob, nW);

  dim3 ggrid(D_ / 128, MROWS / 128);   // (16, 32)
  gemm_bf16_kernel<1, 1><<<ggrid, 256, 0, stream>>>(xb, wqb, nullptr, qb, MROWS, D_, D_);
  gemm_bf16_kernel<1, 1><<<ggrid, 256, 0, stream>>>(xb, wkb, nullptr, kb, MROWS, D_, D_);
  gemm_bf16_kernel<0, 1><<<ggrid, 256, 0, stream>>>(xb, wvb, nullptr, vb, MROWS, D_, D_);

  ld_kernel<<<(MROWS * H_) / 256, 256, 0, stream>>>(x, Wf, ldb);

  attn_kernel<<<2 * H_, 128, 0, stream>>>(qb, kb, vb, ldb, ob);

  rms_kernel<<<MROWS, 256, 0, stream>>>(ob, nw, onb);

  gemm_bf16_kernel<0, 0><<<ggrid, 256, 0, stream>>>(onb, wob, out, nullptr, MROWS, D_, D_);
}
